// GCN_32117765439962
// MI455X (gfx1250) — compile-verified, matrix operand order checked by reading, not run on hardware
//
#include <hip/hip_runtime.h>
#include <hip/hip_bf16.h>

#define F 128

typedef __attribute__((ext_vector_type(16))) __bf16 bf16x16;
typedef __attribute__((ext_vector_type(8)))  float  f32x8;

// ---------------- normalization ----------------

__global__ __launch_bounds__(256) void init_deg(float* deg, int n) {
    int i = blockIdx.x * 256 + threadIdx.x;
    if (i < n) deg[i] = 1.0f;               // implicit self-loop weight 1
}

__global__ __launch_bounds__(256) void accum_deg(float* deg, const int* __restrict__ dst,
                                                 const float* __restrict__ w, int e) {
    int i = blockIdx.x * 256 + threadIdx.x;
    if (i < e) atomicAdd(&deg[dst[i]], w[i]);
}

__global__ __launch_bounds__(256) void finish_norm(float* dinv, float* self_norm, int n) {
    int i = blockIdx.x * 256 + threadIdx.x;
    if (i < n) {
        float di = rsqrtf(dinv[i]);          // deg >= 1 always
        dinv[i] = di;
        self_norm[i] = di * di;
    }
}

__global__ __launch_bounds__(256) void edge_norm_k(const float* __restrict__ dinv,
                                                   const int* __restrict__ src,
                                                   const int* __restrict__ dst,
                                                   const float* __restrict__ w,
                                                   float* __restrict__ en, int e) {
    int i = blockIdx.x * 256 + threadIdx.x;
    if (i < e) en[i] = dinv[src[i]] * w[i] * dinv[dst[i]];
}

// ---------------- weight pre-pack into WMMA B-fragment layout (bf16) ----------------
// tile = kt*8 + nt (kt: K/32 step, nt: N/16 tile). Lane l: col = nt*16 + (l&15),
// K base = kt*32 + (l>>4)*16, 16 consecutive K values -> 16 bf16 (32 bytes/lane).

__global__ __launch_bounds__(256) void pack_w(const float* __restrict__ W, __bf16* __restrict__ Wp) {
    int t = blockIdx.x * 256 + threadIdx.x;
    if (t >= 32 * 32) return;
    int lane = t & 31, tile = t >> 5;
    int kt = tile >> 3, nt = tile & 7;
    int col = nt * 16 + (lane & 15);
    int kb  = kt * 32 + (lane >> 4) * 16;
    __bf16* out = Wp + (size_t)tile * 512 + (size_t)lane * 16;
#pragma unroll
    for (int j = 0; j < 16; ++j)
        out[j] = (__bf16)W[(size_t)(kb + j) * F + col];
}

// ---------------- WMMA GEMM: H[N,128] = act(X)[N,128] @ W[128,128] ----------------
// One wave per 16x128 output tile. Packed bf16 weights staged in LDS (32 KB) per
// block; A half-row (256 B/lane) loaded in one clause; 4x8 = 32 v_wmma per wave.

template <int RELU>
__global__ __launch_bounds__(128) void gemm_wmma(const float* __restrict__ X,
                                                 const __bf16* __restrict__ Wp,
                                                 float* __restrict__ H,
                                                 int nrows) {
    __shared__ __bf16 lds_w[32 * 512];               // 32 KB: full packed weight

    // cooperative stage of Wp into LDS (2048 float4 / 128 threads = 16 each)
    {
        const float4* s4 = (const float4*)Wp;
        float4*       t4 = (float4*)lds_w;
#pragma unroll
        for (int i = 0; i < 16; ++i)
            t4[threadIdx.x + i * 128] = s4[threadIdx.x + i * 128];
    }
    __syncthreads();

    int wave = threadIdx.x >> 5;
    int lane = threadIdx.x & 31;
    int mtile = blockIdx.x * 4 + wave;               // 16-row tile, wave-uniform
    int row0 = mtile * 16;
    if (row0 >= nrows) return;                       // uniform per wave; after barrier

    int half = lane >> 4;                            // 0: K 0..15, 1: K 16..31 of step
    int l15  = lane & 15;
    const float4* xr4 = (const float4*)(X + (size_t)(row0 + l15) * F);

    // Load entire A row (16 x b128 per lane), fuse ReLU, convert to 4 bf16 fragments
    bf16x16 afrag[4];
#pragma unroll
    for (int kt = 0; kt < 4; ++kt) {
        float fv[16];
#pragma unroll
        for (int q = 0; q < 4; ++q)
            *(float4*)(fv + q * 4) = xr4[kt * 8 + half * 4 + q];
#pragma unroll
        for (int j = 0; j < 16; ++j) {
            float v = fv[j];
            if (RELU) v = fmaxf(v, 0.0f);            // compile-time: single v_max
            afrag[kt][j] = (__bf16)v;
        }
    }

    f32x8 acc[8] = {};
#pragma unroll
    for (int kt = 0; kt < 4; ++kt) {
#pragma unroll
        for (int nt = 0; nt < 8; ++nt) {
            bf16x16 b = *(const bf16x16*)(lds_w + ((size_t)(kt * 8 + nt) * 32 + lane) * 16);
            acc[nt] = __builtin_amdgcn_wmma_f32_16x16x32_bf16(
                false, afrag[kt], false, b, (short)0, acc[nt], false, false);
        }
    }

    // C/D layout: VGPR i -> lanes 0..15: M=i, lanes 16..31: M=i+8; N = lane&15
#pragma unroll
    for (int nt = 0; nt < 8; ++nt) {
        int col = nt * 16 + l15;
#pragma unroll
        for (int i = 0; i < 8; ++i) {
            int r = row0 + i + half * 8;
            H[(size_t)r * F + col] = acc[nt][i];
        }
    }
}

// ---------------- aggregation ----------------

__global__ __launch_bounds__(256) void agg_init(const float* __restrict__ h,
                                                const float* __restrict__ self_norm,
                                                const float* __restrict__ bias,
                                                float* __restrict__ agg, int n) {
    size_t idx = (size_t)blockIdx.x * 256 + threadIdx.x;
    if (idx >= (size_t)n * F) return;
    int f = (int)(idx & (F - 1));
    int v = (int)(idx >> 7);
    agg[idx] = self_norm[v] * h[idx] + bias[f];
}

// one wave32 per edge: lane handles float4 -> full 512B row per edge, coalesced
__global__ __launch_bounds__(256) void agg_scatter(const float* __restrict__ h,
                                                   const int* __restrict__ src,
                                                   const int* __restrict__ dst,
                                                   const float* __restrict__ en,
                                                   float* __restrict__ agg, int e) {
    int edge = blockIdx.x * 8 + (threadIdx.x >> 5);
    if (edge >= e) return;
    int lane = threadIdx.x & 31;
    int s = src[edge], d = dst[edge];
    float w = en[edge];
    float4 v = ((const float4*)(h + (size_t)s * F))[lane];
    float* ad = agg + (size_t)d * F + lane * 4;
    atomicAdd(ad + 0, w * v.x);
    atomicAdd(ad + 1, w * v.y);
    atomicAdd(ad + 2, w * v.z);
    atomicAdd(ad + 3, w * v.w);
}

// ---------------- final 128 -> 1 layer ----------------

__global__ __launch_bounds__(256) void final_dot(const float* __restrict__ hpre,
                                                 const float* __restrict__ Wfin,
                                                 float* __restrict__ g, int n) {
    int v = blockIdx.x * 8 + (threadIdx.x >> 5);
    if (v >= n) return;
    int lane = threadIdx.x & 31;
    float4 a = ((const float4*)(hpre + (size_t)v * F))[lane];
    float4 b = ((const float4*)Wfin)[lane];
    float s = fmaxf(a.x, 0.f) * b.x + fmaxf(a.y, 0.f) * b.y +
              fmaxf(a.z, 0.f) * b.z + fmaxf(a.w, 0.f) * b.w;
#pragma unroll
    for (int off = 16; off > 0; off >>= 1)
        s += __shfl_down(s, off, 32);
    if (lane == 0) g[v] = s;
}

__global__ __launch_bounds__(256) void final_init(const float* __restrict__ g,
                                                  const float* __restrict__ self_norm,
                                                  const float* __restrict__ bfin,
                                                  float* __restrict__ out, int n) {
    int i = blockIdx.x * 256 + threadIdx.x;
    if (i < n) out[i] = self_norm[i] * g[i] + bfin[0];
}

__global__ __launch_bounds__(256) void final_scatter(const float* __restrict__ g,
                                                     const int* __restrict__ src,
                                                     const int* __restrict__ dst,
                                                     const float* __restrict__ en,
                                                     float* __restrict__ out, int e) {
    int i = blockIdx.x * 256 + threadIdx.x;
    if (i < e) atomicAdd(&out[dst[i]], en[i] * g[src[i]]);
}

// ---------------- host launcher ----------------

extern "C" void kernel_launch(void* const* d_in, const int* in_sizes, int n_in,
                              void* d_out, int out_size, void* d_ws, size_t ws_size,
                              hipStream_t stream) {
    const float* x     = (const float*)d_in[0];
    const int*   ei    = (const int*)d_in[1];
    const float* w     = (const float*)d_in[2];
    const float* W_in  = (const float*)d_in[3];
    const float* b_in  = (const float*)d_in[4];
    const float* W_mid = (const float*)d_in[5];
    const float* b_mid = (const float*)d_in[6];
    const float* W_fin = (const float*)d_in[7];
    const float* b_fin = (const float*)d_in[8];

    const int N = in_sizes[0] / F;
    const int E = in_sizes[2];
    const int* src = ei;
    const int* dst = ei + E;

    // workspace carve-out
    char* ws = (char*)d_ws;
    size_t o = 0;
    auto carve = [&](size_t bytes) -> void* {
        void* p = ws + o;
        o = (o + bytes + 255) & ~(size_t)255;
        return p;
    };
    float*  dinv      = (float*)carve((size_t)N * 4);
    float*  self_norm = (float*)carve((size_t)N * 4);
    float*  en        = (float*)carve((size_t)E * 4);
    float*  g         = (float*)carve((size_t)N * 4);
    float*  h         = (float*)carve((size_t)N * F * 4);
    float*  agg       = (float*)carve((size_t)N * F * 4);
    __bf16* Wp_in     = (__bf16*)carve((size_t)F * F * 2);
    __bf16* Wp_mid    = (__bf16*)carve((size_t)F * F * 2);

    const int gN  = (N + 255) / 256;
    const int gE  = (E + 255) / 256;
    const int gNF = (int)(((size_t)N * F + 255) / 256);
    const int gEw = (E + 7) / 8;                 // wave-per-edge
    const int gNw = (N + 7) / 8;                 // wave-per-node
    const int mt  = (N + 15) / 16;
    const int gG  = (mt + 3) / 4;                // 4 waves / block

    // normalization
    init_deg<<<gN, 256, 0, stream>>>(dinv, N);
    accum_deg<<<gE, 256, 0, stream>>>(dinv, dst, w, E);
    finish_norm<<<gN, 256, 0, stream>>>(dinv, self_norm, N);
    edge_norm_k<<<gE, 256, 0, stream>>>(dinv, src, dst, w, en, E);

    // pack weights once (bf16, WMMA B-frag layout)
    pack_w<<<4, 256, 0, stream>>>(W_in, Wp_in);
    pack_w<<<4, 256, 0, stream>>>(W_mid, Wp_mid);

    // layer 1: x -> agg (pre-ReLU)
    gemm_wmma<0><<<gG, 128, 0, stream>>>(x, Wp_in, h, N);
    agg_init<<<gNF, 256, 0, stream>>>(h, self_norm, b_in, agg, N);
    agg_scatter<<<gEw, 256, 0, stream>>>(h, src, dst, en, agg, E);

    // layer 2: ReLU fused into GEMM load
    gemm_wmma<1><<<gG, 128, 0, stream>>>(agg, Wp_mid, h, N);
    agg_init<<<gNF, 256, 0, stream>>>(h, self_norm, b_mid, agg, N);
    agg_scatter<<<gEw, 256, 0, stream>>>(h, src, dst, en, agg, E);

    // layer 3
    gemm_wmma<1><<<gG, 128, 0, stream>>>(agg, Wp_mid, h, N);
    agg_init<<<gNF, 256, 0, stream>>>(h, self_norm, b_mid, agg, N);
    agg_scatter<<<gEw, 256, 0, stream>>>(h, src, dst, en, agg, E);

    // final 128->1 layer (ReLU fused into dot)
    final_dot<<<gNw, 256, 0, stream>>>(agg, W_fin, g, N);
    final_init<<<gN, 256, 0, stream>>>(g, self_norm, b_fin, (float*)d_out, N);
    final_scatter<<<gE, 256, 0, stream>>>(g, src, dst, en, (float*)d_out, E);
}
